// enc_graph_6236292514562
// MI455X (gfx1250) — matrix operand order repaired and checked by reading, not compiled, not run on hardware
//
#include <hip/hip_runtime.h>

typedef __attribute__((ext_vector_type(16))) __bf16 v16bf;
typedef __attribute__((ext_vector_type(8)))  float  v8f;

#define BATCH 1024
#define NATOM 60
#define DMAX  5
#define FA    62
#define FB    6
#define WID   128
#define ROWS  (BATCH*NATOM)       // 61440
#define KP1   96                  // Fin1 = 68 -> pad 96  (3 k-steps)
#define KP23  160                 // Fin2/3 = 134 -> pad 160 (5 k-steps)
#define KSZ   16
#define KCONV (KSZ*WID)           // 2048
#define TCONV (NATOM-KSZ+1)       // 45
#define CROWS (BATCH*TCONV)       // 46080
#define EPSBN 1e-3f

__device__ __forceinline__ unsigned short f2bf(float f) {
  unsigned u = __float_as_uint(f);
  u += 0x7FFFu + ((u >> 16) & 1u);          // round-to-nearest-even
  return (unsigned short)(u >> 16);
}
__device__ __forceinline__ float bf2f(unsigned short h) {
  return __uint_as_float(((unsigned)h) << 16);
}
__device__ __forceinline__ v16bf pack16(uint4 lo, uint4 hi) {
  v16bf v;
  __builtin_memcpy(&v, &lo, 16);
  __builtin_memcpy(reinterpret_cast<char*>(&v) + 16, &hi, 16);
  return v;
}

// ---------------- prep kernels ----------------

__global__ void deg_kernel(const int* __restrict__ edges, int* __restrict__ deg) {
  int i = blockIdx.x * blockDim.x + threadIdx.x;
  if (i >= ROWS) return;
  int c = 0;
#pragma unroll
  for (int d = 0; d < DMAX; ++d) c += (edges[i * DMAX + d] >= 0);
  deg[i] = c;
}

// W [6, Fin, 128] f32 -> Wt [6, 128, Kpad] bf16 (K-contiguous per output column)
__global__ void prep_ngf_w(const float* __restrict__ W, unsigned short* __restrict__ Wt,
                           int Fin, int Kpad) {
  int idx = blockIdx.x * blockDim.x + threadIdx.x;
  int total = (DMAX + 1) * WID * Kpad;
  if (idx >= total) return;
  int k   = idx % Kpad;
  int col = (idx / Kpad) % WID;
  int d   = idx / (Kpad * WID);
  float v = (k < Fin) ? W[((size_t)d * Fin + k) * WID + col] : 0.f;
  Wt[idx] = f2bf(v);
}

// Wc [2048, 128] f32 -> Wct [128, 2048] bf16
__global__ void prep_conv_w(const float* __restrict__ Wc, unsigned short* __restrict__ Wct) {
  int idx = blockIdx.x * blockDim.x + threadIdx.x;
  if (idx >= WID * KCONV) return;
  int k = idx % KCONV;
  int f = idx / KCONV;
  Wct[idx] = f2bf(Wc[(size_t)k * WID + f]);
}

// ---------------- feature builders (gather + concat, bf16 out) ----------------

__global__ void feats1_kernel(const float* __restrict__ atoms,
                              const float* __restrict__ bonds,
                              const int* __restrict__ edges,
                              unsigned short* __restrict__ feats) {
  int row = blockIdx.x;              // b*N + n
  int b   = row / NATOM;
  int k   = threadIdx.x;             // 0..KP1-1
  float val = 0.f;
  if (k < FA) {
    val = atoms[(size_t)row * FA + k];
#pragma unroll
    for (int d = 0; d < DMAX; ++d) {
      int e = edges[row * DMAX + d];
      if (e >= 0) val += atoms[((size_t)b * NATOM + e) * FA + k];
    }
  } else if (k < FA + FB) {
    int j = k - FA;
#pragma unroll
    for (int d = 0; d < DMAX; ++d) val += bonds[((size_t)row * DMAX + d) * FB + j];
  }
  feats[(size_t)row * KP1 + k] = f2bf(val);
}

__global__ void feats23_kernel(const unsigned short* __restrict__ x,  // [ROWS,128] bf16
                               const float* __restrict__ bonds,
                               const int* __restrict__ edges,
                               unsigned short* __restrict__ feats) {
  int row = blockIdx.x;
  int b   = row / NATOM;
  int k   = threadIdx.x;             // 0..KP23-1
  float val = 0.f;
  if (k < WID) {
    val = bf2f(x[(size_t)row * WID + k]);
#pragma unroll
    for (int d = 0; d < DMAX; ++d) {
      int e = edges[row * DMAX + d];
      if (e >= 0) val += bf2f(x[((size_t)b * NATOM + e) * WID + k]);
    }
  } else if (k < WID + FB) {
    int j = k - WID;
#pragma unroll
    for (int d = 0; d < DMAX; ++d) val += bonds[((size_t)row * DMAX + d) * FB + j];
  }
  feats[(size_t)row * KP23 + k] = f2bf(val);
}

// ---------------- NGF per-degree GEMM (WMMA, masked-by-degree) ----------------
// Each wave owns 32 rows x 128 cols: two A tiles share every B tile (2 wmma / B-load).

__global__ void __launch_bounds__(128) ngf_gemm_kernel(
    const unsigned short* __restrict__ feats,  // [ROWS, Kpad] bf16
    const unsigned short* __restrict__ Wt,     // [6, 128, Kpad] bf16
    const int* __restrict__ deg,               // [ROWS]
    const float* __restrict__ bias,            // [6, 128]
    const float* __restrict__ g,  const float* __restrict__ be,
    const float* __restrict__ mu, const float* __restrict__ va,
    unsigned short* __restrict__ xout,         // [ROWS, 128] bf16
    int Kpad) {
  const int lane    = threadIdx.x & 31;
  const int wv      = threadIdx.x >> 5;
  const int rowbase = (blockIdx.x * 4 + wv) * 32;
  const int c16     = lane & 15;
  const int hi      = lane >> 4;               // 0 or 1
  const int half16  = hi * 16;
  const int lrow0   = rowbase + c16;
  const int lrow1   = rowbase + 16 + c16;
  const int deg0    = deg[lrow0];
  const int deg1    = deg[lrow1];

  v8f zero = {0.f, 0.f, 0.f, 0.f, 0.f, 0.f, 0.f, 0.f};
  v8f acc0[8], acc1[8];
#pragma unroll
  for (int t = 0; t < 8; ++t) { acc0[t] = zero; acc1[t] = zero; }

  uint4 z4 = make_uint4(0u, 0u, 0u, 0u);
  const int nk = Kpad >> 5;
  const unsigned short* arow0 = feats + (size_t)lrow0 * Kpad + (hi ? 8 : 0);
  const unsigned short* arow1 = feats + (size_t)lrow1 * Kpad + (hi ? 8 : 0);

  for (int kt = 0; kt < nk; ++kt) {
    const unsigned short* ap0 = arow0 + kt * 32;
    const unsigned short* ap1 = arow1 + kt * 32;
    uint4 a00 = *(const uint4*)(ap0);
    uint4 a01 = *(const uint4*)(ap0 + 16);
    uint4 a10 = *(const uint4*)(ap1);
    uint4 a11 = *(const uint4*)(ap1 + 16);
    for (int d = 0; d <= DMAX; ++d) {
      unsigned pres = __builtin_amdgcn_ballot_w32((deg0 == d) || (deg1 == d));
      if (pres == 0u) continue;                // wave-uniform skip, EXEC untouched
      bool k0 = (deg0 == d), k1 = (deg1 == d);
      v16bf av0 = pack16(k0 ? a00 : z4, k0 ? a01 : z4);
      v16bf av1 = pack16(k1 ? a10 : z4, k1 ? a11 : z4);
      const unsigned short* wd =
          Wt + ((size_t)d * WID + c16) * Kpad + kt * 32 + hi * 8;
#pragma unroll
      for (int t = 0; t < 8; ++t) {
        const unsigned short* bp = wd + (size_t)t * 16 * Kpad;
        v16bf bv = pack16(*(const uint4*)bp, *(const uint4*)(bp + 16));
        acc0[t] = __builtin_amdgcn_wmma_f32_16x16x32_bf16(
            false, av0, false, bv, (short)0, acc0[t], false, false);
        acc1[t] = __builtin_amdgcn_wmma_f32_16x16x32_bf16(
            false, av1, false, bv, (short)0, acc1[t], false, false);
      }
    }
  }

  int dg0[8], dg1[8];
#pragma unroll
  for (int r = 0; r < 8; ++r) {
    dg0[r] = deg[rowbase + r + hi * 8];
    dg1[r] = deg[rowbase + 16 + r + hi * 8];
  }
#pragma unroll
  for (int t = 0; t < 8; ++t) {
    int col = t * 16 + c16;
    float s = g[col] * rsqrtf(va[col] + EPSBN);
    float o = be[col] - s * mu[col];
#pragma unroll
    for (int r = 0; r < 8; ++r) {
      int rw0 = rowbase + r + hi * 8;
      float v = acc0[t][r] + bias[dg0[r] * WID + col];
      v = fmaf(v, s, o);
      v = v > 0.f ? v : 0.f;
      xout[(size_t)rw0 * WID + col] = f2bf(v);

      int rw1 = rowbase + 16 + r + hi * 8;
      float w = acc1[t][r] + bias[dg1[r] * WID + col];
      w = fmaf(w, s, o);
      w = w > 0.f ? w : 0.f;
      xout[(size_t)rw1 * WID + col] = f2bf(w);
    }
  }
}

// ---------------- Conv1D as implicit GEMM (K=2048 contiguous patches) ----------------
// 32 output rows per wave: each B tile (weight slice) feeds two wmma.

__global__ void __launch_bounds__(128) conv_gemm_kernel(
    const unsigned short* __restrict__ x,    // [B, 60, 128] bf16
    const unsigned short* __restrict__ Wct,  // [128, 2048] bf16
    const float* __restrict__ g,  const float* __restrict__ be,
    const float* __restrict__ mu, const float* __restrict__ va,
    float* __restrict__ out) {               // [B, 45, 128] f32
  const int lane    = threadIdx.x & 31;
  const int wv      = threadIdx.x >> 5;
  const int rowbase = (blockIdx.x * 4 + wv) * 32;
  const int c16     = lane & 15;
  const int hi      = lane >> 4;
  const int half16  = hi * 16;

  int rowid0 = rowbase + c16;
  int rowid1 = rowbase + 16 + c16;
  int b0 = rowid0 / TCONV, t00 = rowid0 % TCONV;
  int b1 = rowid1 / TCONV, t01 = rowid1 % TCONV;
  const unsigned short* arow0 = x + ((size_t)b0 * NATOM + t00) * WID + (hi ? 8 : 0);
  const unsigned short* arow1 = x + ((size_t)b1 * NATOM + t01) * WID + (hi ? 8 : 0);

  v8f zero = {0.f, 0.f, 0.f, 0.f, 0.f, 0.f, 0.f, 0.f};
  v8f acc0[8], acc1[8];
#pragma unroll
  for (int t = 0; t < 8; ++t) { acc0[t] = zero; acc1[t] = zero; }

  for (int kt = 0; kt < (KCONV >> 5); ++kt) {
    const unsigned short* ap0 = arow0 + kt * 32;
    const unsigned short* ap1 = arow1 + kt * 32;
    v16bf av0 = pack16(*(const uint4*)(ap0), *(const uint4*)(ap0 + 16));
    v16bf av1 = pack16(*(const uint4*)(ap1), *(const uint4*)(ap1 + 16));
    const unsigned short* wb = Wct + (size_t)c16 * KCONV + kt * 32 + hi * 8;
#pragma unroll
    for (int t = 0; t < 8; ++t) {
      const unsigned short* bp = wb + (size_t)t * 16 * KCONV;
      v16bf bv = pack16(*(const uint4*)bp, *(const uint4*)(bp + 16));
      acc0[t] = __builtin_amdgcn_wmma_f32_16x16x32_bf16(
          false, av0, false, bv, (short)0, acc0[t], false, false);
      acc1[t] = __builtin_amdgcn_wmma_f32_16x16x32_bf16(
          false, av1, false, bv, (short)0, acc1[t], false, false);
    }
  }

#pragma unroll
  for (int t = 0; t < 8; ++t) {
    int col = t * 16 + c16;
    float s = g[col] * rsqrtf(va[col] + EPSBN);
    float o = be[col] - s * mu[col];
#pragma unroll
    for (int r = 0; r < 8; ++r) {
      int rw0 = rowbase + r + hi * 8;
      float v = fmaf(acc0[t][r], s, o);
      out[(size_t)rw0 * WID + col] = v > 0.f ? v : 0.f;

      int rw1 = rowbase + 16 + r + hi * 8;
      float w = fmaf(acc1[t][r], s, o);
      out[(size_t)rw1 * WID + col] = w > 0.f ? w : 0.f;
    }
  }
}

// ---------------- launch ----------------

extern "C" void kernel_launch(void* const* d_in, const int* in_sizes, int n_in,
                              void* d_out, int out_size, void* d_ws, size_t ws_size,
                              hipStream_t stream) {
  (void)in_sizes; (void)n_in; (void)out_size; (void)ws_size;
  const float* atoms = (const float*)d_in[0];
  const float* bonds = (const float*)d_in[1];
  const int*   edges = (const int*)  d_in[2];
  const float* W1 = (const float*)d_in[3]; const float* b1 = (const float*)d_in[4];
  const float* W2 = (const float*)d_in[5]; const float* b2 = (const float*)d_in[6];
  const float* W3 = (const float*)d_in[7]; const float* b3 = (const float*)d_in[8];
  const float* Wc = (const float*)d_in[9];
  const float* bn[4][4];
  for (int i = 0; i < 4; ++i)
    for (int j = 0; j < 4; ++j) bn[i][j] = (const float*)d_in[10 + i * 4 + j];

  char* w = (char*)d_ws;
  size_t off = 0;
  auto alloc = [&](size_t bytes) -> void* {
    void* p = w + off;
    off = (off + bytes + 255) & ~(size_t)255;
    return p;
  };
  int*            dDeg = (int*)alloc((size_t)ROWS * sizeof(int));
  unsigned short* dF   = (unsigned short*)alloc((size_t)ROWS * KP23 * 2);
  unsigned short* dX   = (unsigned short*)alloc((size_t)ROWS * WID * 2);
  unsigned short* dWt1 = (unsigned short*)alloc((size_t)6 * WID * KP1 * 2);
  unsigned short* dWt2 = (unsigned short*)alloc((size_t)6 * WID * KP23 * 2);
  unsigned short* dWt3 = (unsigned short*)alloc((size_t)6 * WID * KP23 * 2);
  unsigned short* dWct = (unsigned short*)alloc((size_t)WID * KCONV * 2);

  deg_kernel<<<(ROWS + 255) / 256, 256, 0, stream>>>(edges, dDeg);
  { int tot = 6 * WID * KP1;
    prep_ngf_w<<<(tot + 255) / 256, 256, 0, stream>>>(W1, dWt1, FA + FB, KP1); }
  { int tot = 6 * WID * KP23;
    prep_ngf_w<<<(tot + 255) / 256, 256, 0, stream>>>(W2, dWt2, WID + FB, KP23); }
  { int tot = 6 * WID * KP23;
    prep_ngf_w<<<(tot + 255) / 256, 256, 0, stream>>>(W3, dWt3, WID + FB, KP23); }
  { int tot = WID * KCONV;
    prep_conv_w<<<(tot + 255) / 256, 256, 0, stream>>>(Wc, dWct); }

  // layer 1
  feats1_kernel<<<ROWS, KP1, 0, stream>>>(atoms, bonds, edges, dF);
  ngf_gemm_kernel<<<ROWS / 128, 128, 0, stream>>>(dF, dWt1, dDeg, b1,
      bn[0][0], bn[0][1], bn[0][2], bn[0][3], dX, KP1);
  // layer 2
  feats23_kernel<<<ROWS, KP23, 0, stream>>>(dX, bonds, edges, dF);
  ngf_gemm_kernel<<<ROWS / 128, 128, 0, stream>>>(dF, dWt2, dDeg, b2,
      bn[1][0], bn[1][1], bn[1][2], bn[1][3], dX, KP23);
  // layer 3
  feats23_kernel<<<ROWS, KP23, 0, stream>>>(dX, bonds, edges, dF);
  ngf_gemm_kernel<<<ROWS / 128, 128, 0, stream>>>(dF, dWt3, dDeg, b3,
      bn[2][0], bn[2][1], bn[2][2], bn[2][3], dX, KP23);
  // conv + bn4 + relu
  conv_gemm_kernel<<<CROWS / 128, 128, 0, stream>>>(dX, dWct,
      bn[3][0], bn[3][1], bn[3][2], bn[3][3], (float*)d_out);
}
